// MHA_37048387895985
// MI455X (gfx1250) — matrix operand order repaired and checked by reading, not compiled, not run on hardware
//
#include <hip/hip_runtime.h>
#include <hip/hip_bf16.h>

// ---------------------------------------------------------------------------
// Linformer-style linear attention on gfx1250 (MI455X) via bf16 WMMA.
// Round 4: 64x128 block tile, wave tile 32x32 -> 4 WMMAs/wave/K-step from
// 2 A-frags x 2 B-frags (both operands reused twice); double-buffered LDS
// with register-staged global loads (one barrier per K-step); fragment-ready
// LDS layout so fragments load as ds_load_b128.
// All GEMM dims are multiples of the 64x128x32 block tile: no bounds checks,
// EXEC stays all-ones around WMMA (ISA requirement).
// ---------------------------------------------------------------------------

typedef __attribute__((ext_vector_type(16))) __bf16 v16bf;
typedef __attribute__((ext_vector_type(8)))  float  v8f;

#define BM 64
#define BN 128
#define BK 32
#define NT 256            // 8 waves of 32
#define FRAG_U 12         // uints per lane-row (8 used + pad, 48B: 16B-aligned)

__device__ __forceinline__ unsigned short f32_to_bf16(float f) {
    unsigned int u = __float_as_uint(f);
    u += 0x7FFFu + ((u >> 16) & 1u);          // round-to-nearest-even
    return (unsigned short)(u >> 16);
}
__device__ __forceinline__ float bf16_to_f32(unsigned short s) {
    return __uint_as_float(((unsigned int)s) << 16);
}
__device__ __forceinline__ unsigned short to_bf16_bits(float x)          { return f32_to_bf16(x); }
__device__ __forceinline__ unsigned short to_bf16_bits(unsigned short x) { return x; }
__device__ __forceinline__ void store_c(float* p, float v)          { *p = v; }
__device__ __forceinline__ void store_c(unsigned short* p, float v) { *p = f32_to_bf16(v); }

// Vector-of-4 global load -> 4 bf16 bit patterns
__device__ __forceinline__ void load4_bf16(const float* p, unsigned short o[4]) {
    float4 t = *(const float4*)p;
    o[0] = f32_to_bf16(t.x); o[1] = f32_to_bf16(t.y);
    o[2] = f32_to_bf16(t.z); o[3] = f32_to_bf16(t.w);
}
__device__ __forceinline__ void load4_bf16(const unsigned short* p, unsigned short o[4]) {
    ushort4 t = *(const ushort4*)p;
    o[0] = t.x; o[1] = t.y; o[2] = t.z; o[3] = t.w;
}

// WMMA fragment position maps (CDNA5 wave32 layouts, CDNA5 ISA):
//  A 16x32:  lane = (m&15) + 16*((k>>3)&1),  slot = (k&7) + 8*(k>>4)
//  B 32x16:  lane = (n&15) + 16*((k>>3)&1),  slot = (k&7) + 8*(k>>4)

// EPI: 0 = plain store, 1 = elu(x)+1 (== x>0 ? x+1 : exp(x)), 2 = divide by qz[row]
// VA/VB: operand has unit column stride & 16B-aligned tiles -> vector-4 staging
template<typename TA, typename TB, typename TC, int EPI, bool VA, bool VB>
__global__ void __launch_bounds__(NT)
gemm_bf16_wmma(const TA* __restrict__ A, const TB* __restrict__ B,
               TC* __restrict__ C, const float* __restrict__ divRow,
               int M, int N, int Kd,
               long aRow, long aCol, long bRow, long bCol, long cRow, long cCol,
               long aBO, long aBI, long bBO, long bBI, long cBO, long cBI,
               int innerCount)
{
    // Double-buffered fragment-ready tiles: [buf][subtile][lane][FRAG_U uints]
    __shared__ unsigned int As4[2][4][32][FRAG_U];   // 4 M-subtiles  (12 KB)
    __shared__ unsigned int Bs4[2][8][32][FRAG_U];   // 8 N-subtiles  (24 KB)

    const int tid  = threadIdx.x;
    const int lane = tid & 31;
    const int wid  = tid >> 5;      // 0..7
    const int wm   = wid >> 2;      // 0..1  (pair of M sub-tiles: 2*wm, 2*wm+1)
    const int wn   = wid & 3;       // 0..3  (pair of N sub-tiles: 2*wn, 2*wn+1)

    const int  z  = blockIdx.z;
    const int  zo = z / innerCount;
    const int  zi = z % innerCount;

    const TA* Ab = A + (long)zo * aBO + (long)zi * aBI + (long)(blockIdx.y * BM) * aRow;
    const TB* Bb = B + (long)zo * bBO + (long)zi * bBI + (long)(blockIdx.x * BN) * bCol;
    TC*       Cb = C + (long)zo * cBO + (long)zi * cBI
                     + (long)(blockIdx.y * BM) * cRow + (long)(blockIdx.x * BN) * cCol;

    // Register staging for the software pipeline
    unsigned short hA[8][4];   // VA: [0..1][*] used; scalar: [i][0] per element
    unsigned short hB[16][4];  // VB: [0..3][*]; scalar: [i][0] per element

    auto loadA = [&](int k0) {
        if (VA) {
            #pragma unroll
            for (int i = 0; i < 2; ++i) {           // 512 groups of 4
                int g  = tid + i * NT;
                int m  = g >> 3;
                int k4 = (g & 7) << 2;
                load4_bf16(Ab + (long)m * aRow + (k0 + k4), hA[i]);
            }
        } else {
            #pragma unroll
            for (int i = 0; i < 8; ++i) {           // 2048 scalar elements
                int e = tid + i * NT;
                int m = e >> 5, k = e & 31;
                hA[i][0] = to_bf16_bits(Ab[(long)m * aRow + (long)(k0 + k) * aCol]);
            }
        }
    };
    auto storeA = [&](int buf) {
        if (VA) {
            #pragma unroll
            for (int i = 0; i < 2; ++i) {
                int g  = tid + i * NT;
                int m  = g >> 3;
                int k4 = (g & 7) << 2;
                int sub  = m >> 4;
                int lrow = (m & 15) + 16 * ((k4 >> 3) & 1);
                int idx  = (k4 & 7) + 8 * (k4 >> 4);      // multiple of 4
                uint2 pv;
                pv.x = (unsigned int)hA[i][0] | ((unsigned int)hA[i][1] << 16);
                pv.y = (unsigned int)hA[i][2] | ((unsigned int)hA[i][3] << 16);
                *(uint2*)&As4[buf][sub][lrow][idx >> 1] = pv;
            }
        } else {
            #pragma unroll
            for (int i = 0; i < 8; ++i) {
                int e = tid + i * NT;
                int m = e >> 5, k = e & 31;
                int sub  = m >> 4;
                int lrow = (m & 15) + 16 * ((k >> 3) & 1);
                int idx  = (k & 7) + 8 * (k >> 4);
                ((unsigned short*)&As4[buf][sub][lrow][0])[idx] = hA[i][0];
            }
        }
    };
    auto loadB = [&](int k0) {
        if (VB) {
            #pragma unroll
            for (int i = 0; i < 4; ++i) {           // 1024 groups of 4
                int g  = tid + i * NT;
                int k  = g >> 5;
                int n4 = (g & 31) << 2;
                load4_bf16(Bb + (long)(k0 + k) * bRow + n4, hB[i]);
            }
        } else {
            #pragma unroll
            for (int i = 0; i < 16; ++i) {          // 4096 scalar elements
                int e = tid + i * NT;
                int k = e >> 7, n = e & 127;
                hB[i][0] = to_bf16_bits(Bb[(long)(k0 + k) * bRow + (long)n * bCol]);
            }
        }
    };
    auto storeB = [&](int buf) {
        if (VB) {
            #pragma unroll
            for (int i = 0; i < 4; ++i) {
                int g  = tid + i * NT;
                int k  = g >> 5;
                int n4 = (g & 31) << 2;
                int sub = n4 >> 4;
                int idx = (k & 7) + 8 * (k >> 4);
                unsigned short* bh = (unsigned short*)&Bs4[buf][sub][0][0];
                #pragma unroll
                for (int j = 0; j < 4; ++j) {
                    int lrow = ((n4 + j) & 15) + 16 * ((k >> 3) & 1);
                    bh[lrow * (FRAG_U * 2) + idx] = hB[i][j];
                }
            }
        } else {
            #pragma unroll
            for (int i = 0; i < 16; ++i) {
                int e = tid + i * NT;
                int k = e >> 7, n = e & 127;
                int sub  = n >> 4;
                int lrow = (n & 15) + 16 * ((k >> 3) & 1);
                int idx  = (k & 7) + 8 * (k >> 4);
                ((unsigned short*)&Bs4[buf][sub][lrow][0])[idx] = hB[i][0];
            }
        }
    };

    v8f acc00 = {}, acc01 = {}, acc10 = {}, acc11 = {};

    const int nIter = Kd / BK;

    // Prologue: fill buffer 0
    loadA(0); loadB(0);
    storeA(0); storeB(0);
    __syncthreads();

    for (int it = 0; it < nIter; ++it) {
        const int cur = it & 1;

        // Issue next tile's global loads early (latency hidden behind WMMA)
        if (it + 1 < nIter) { loadA((it + 1) * BK); loadB((it + 1) * BK); }

        // Fragments: 2x b128 per fragment; 4 WMMAs reuse both A and B frags
        union { v16bf v; uint4 q[2]; } a0, a1, b0, b1;
        a0.q[0] = *(const uint4*)&As4[cur][2 * wm + 0][lane][0];
        a0.q[1] = *(const uint4*)&As4[cur][2 * wm + 0][lane][4];
        a1.q[0] = *(const uint4*)&As4[cur][2 * wm + 1][lane][0];
        a1.q[1] = *(const uint4*)&As4[cur][2 * wm + 1][lane][4];
        b0.q[0] = *(const uint4*)&Bs4[cur][2 * wn + 0][lane][0];
        b0.q[1] = *(const uint4*)&Bs4[cur][2 * wn + 0][lane][4];
        b1.q[0] = *(const uint4*)&Bs4[cur][2 * wn + 1][lane][0];
        b1.q[1] = *(const uint4*)&Bs4[cur][2 * wn + 1][lane][4];

        acc00 = __builtin_amdgcn_wmma_f32_16x16x32_bf16(
                    false, a0.v, false, b0.v, (short)0, acc00, false, false);
        acc01 = __builtin_amdgcn_wmma_f32_16x16x32_bf16(
                    false, a0.v, false, b1.v, (short)0, acc01, false, false);
        acc10 = __builtin_amdgcn_wmma_f32_16x16x32_bf16(
                    false, a1.v, false, b0.v, (short)0, acc10, false, false);
        acc11 = __builtin_amdgcn_wmma_f32_16x16x32_bf16(
                    false, a1.v, false, b1.v, (short)0, acc11, false, false);

        // Store next tile into the other buffer; WAR on buf[cur] is fenced by
        // the end-of-iteration barrier.
        if (it + 1 < nIter) { storeA(1 - cur); storeB(1 - cur); }
        __syncthreads();
    }

    // Epilogue: C layout — lane<16: N=lane, M=r; lane>=16: N=lane-16, M=8+r
    const int rhi = (lane < 16) ? 0 : 8;
    const v8f* accs[2][2] = { { &acc00, &acc01 }, { &acc10, &acc11 } };
    #pragma unroll
    for (int tm = 0; tm < 2; ++tm) {
        #pragma unroll
        for (int tn = 0; tn < 2; ++tn) {
            const v8f& acc = *accs[tm][tn];
            const int  col = (2 * wn + tn) * 16 + (lane & 15);
            #pragma unroll
            for (int r = 0; r < 8; ++r) {
                int   row = (2 * wm + tm) * 16 + r + rhi;
                float v   = acc[r];
                if (EPI == 1) v = (v > 0.0f) ? (v + 1.0f) : __expf(v);
                if (EPI == 2) {
                    long grow = (long)blockIdx.y * BM + row;
                    v = v / divRow[(long)z * M + grow];
                }
                store_c(&Cb[(long)row * cRow + (long)col * cCol], v);
            }
        }
    }
}

// z[b, nh] = sum_{k<Kr} phi_k[b, k, nh]
__global__ void __launch_bounds__(256)
reduce_z(const unsigned short* __restrict__ phiK, float* __restrict__ z,
         int Kr, int NH)
{
    int j  = blockIdx.x * blockDim.x + threadIdx.x;   // b*NH + nh
    int b  = j / NH, nh = j % NH;
    const unsigned short* p = phiK + (long)b * Kr * NH + nh;
    float s = 0.0f;
    for (int k = 0; k < Kr; ++k) s += bf16_to_f32(p[(long)k * NH]);
    z[j] = s;
}

// qz[(b*N+n)*S + q] = eps + sum_h phi_q[b,q,n*H+h] * z[b,n*H+h]   (wave per row)
__global__ void __launch_bounds__(256)
compute_qz(const unsigned short* __restrict__ phiQ, const float* __restrict__ z,
           float* __restrict__ qz, int S, int Nh, int H)
{
    const int NH  = Nh * H;
    long gw   = (long)blockIdx.x * (blockDim.x >> 5) + (threadIdx.x >> 5);
    int  lane = threadIdx.x & 31;
    long q  = gw % S;
    long bn = gw / S;
    int  n  = (int)(bn % Nh);
    int  b  = (int)(bn / Nh);
    const unsigned short* pq = phiQ + ((long)b * S + q) * NH + (long)n * H;
    const float*          pz = z    + (long)b * NH + (long)n * H;
    float s = 0.0f;
    for (int h = lane; h < H; h += 32) s += bf16_to_f32(pq[h]) * pz[h];
    #pragma unroll
    for (int off = 16; off; off >>= 1) s += __shfl_xor(s, off, 32);
    if (lane == 0) qz[gw] = s + 1e-6f;
}

extern "C" void kernel_launch(void* const* d_in, const int* in_sizes, int n_in,
                              void* d_out, int out_size, void* d_ws, size_t ws_size,
                              hipStream_t stream)
{
    (void)in_sizes; (void)n_in; (void)out_size; (void)ws_size;
    const float* q  = (const float*)d_in[0];
    const float* k  = (const float*)d_in[1];
    const float* v  = (const float*)d_in[2];
    const float* Wq = (const float*)d_in[3];
    const float* Wk = (const float*)d_in[4];
    const float* Wv = (const float*)d_in[5];
    const float* kd = (const float*)d_in[6];
    const float* vd = (const float*)d_in[7];
    float* out = (float*)d_out;

    constexpr int Bc = 32, S = 512, D = 2048, Nh = 16, H = 128, Kr = 64;
    constexpr int NH = Nh * H;   // 2048

    // Workspace carve-out (all fully written before read; ~131 MB)
    char*  w   = (char*)d_ws;
    size_t off = 0;
    auto alloc = [&](size_t bytes) -> char* {
        char* p = w + off;
        off = (off + bytes + 255) & ~(size_t)255;
        return p;
    };
    unsigned short* key_ds = (unsigned short*)alloc((size_t)Bc * Kr * D  * 2);
    unsigned short* val_ds = (unsigned short*)alloc((size_t)Bc * Kr * D  * 2);
    unsigned short* phi_k  = (unsigned short*)alloc((size_t)Bc * Kr * NH * 2);
    unsigned short* vals   = (unsigned short*)alloc((size_t)Bc * Kr * NH * 2);
    unsigned short* phi_q  = (unsigned short*)alloc((size_t)Bc * S  * NH * 2);
    float*          sMat   = (float*)alloc((size_t)Bc * Nh * H * H * 4);
    float*          zVec   = (float*)alloc((size_t)Bc * NH * 4);
    float*          qzBuf  = (float*)alloc((size_t)Bc * Nh * S * 4);

    dim3 blk(NT);

    // 1) key_ds[b] = key_down (64x512) @ k[b] (512x2048)
    gemm_bf16_wmma<float, float, unsigned short, 0, true, true>
        <<<dim3(D / BN, Kr / BM, Bc), blk, 0, stream>>>(
            kd, k, key_ds, nullptr, Kr, D, S,
            (long)S, 1L, (long)D, 1L, (long)D, 1L,
            0L, 0L, (long)S * D, 0L, (long)Kr * D, 0L, 1);

    // 2) val_ds[b] = value_down @ v[b]
    gemm_bf16_wmma<float, float, unsigned short, 0, true, true>
        <<<dim3(D / BN, Kr / BM, Bc), blk, 0, stream>>>(
            vd, v, val_ds, nullptr, Kr, D, S,
            (long)S, 1L, (long)D, 1L, (long)D, 1L,
            0L, 0L, (long)S * D, 0L, (long)Kr * D, 0L, 1);

    // 3) phi_k[b] = elu1(key_ds[b] (64x2048) @ Wk (2048x2048))
    gemm_bf16_wmma<unsigned short, float, unsigned short, 1, true, true>
        <<<dim3(NH / BN, Kr / BM, Bc), blk, 0, stream>>>(
            key_ds, Wk, phi_k, nullptr, Kr, NH, D,
            (long)D, 1L, (long)NH, 1L, (long)NH, 1L,
            (long)Kr * D, 0L, 0L, 0L, (long)Kr * NH, 0L, 1);

    // 4) values[b] = val_ds[b] @ Wv
    gemm_bf16_wmma<unsigned short, float, unsigned short, 0, true, true>
        <<<dim3(NH / BN, Kr / BM, Bc), blk, 0, stream>>>(
            val_ds, Wv, vals, nullptr, Kr, NH, D,
            (long)D, 1L, (long)NH, 1L, (long)NH, 1L,
            (long)Kr * D, 0L, 0L, 0L, (long)Kr * NH, 0L, 1);

    // 5) phi_q[b] = elu1(q[b] (512x2048) @ Wq)   — the 137 GFLOP GEMM
    gemm_bf16_wmma<float, float, unsigned short, 1, true, true>
        <<<dim3(NH / BN, S / BM, Bc), blk, 0, stream>>>(
            q, Wq, phi_q, nullptr, S, NH, D,
            (long)D, 1L, (long)NH, 1L, (long)NH, 1L,
            (long)S * D, 0L, 0L, 0L, (long)S * NH, 0L, 1);

    // 6) z[b,nh] = sum_k phi_k
    reduce_z<<<dim3((Bc * NH) / 256), blk, 0, stream>>>(phi_k, zVec, Kr, NH);

    // 7) s[b,n] = phi_k[b,:,n,:]^T (128x64) @ values[b,:,n,:] (64x128)
    //    A is a transposed view (col stride NH) -> scalar A staging
    gemm_bf16_wmma<unsigned short, unsigned short, float, 0, false, true>
        <<<dim3(H / BN, H / BM, Bc * Nh), blk, 0, stream>>>(
            phi_k, vals, sMat, nullptr, H, H, Kr,
            1L, (long)NH, (long)NH, 1L, (long)H, 1L,
            (long)Kr * NH, (long)H, (long)Kr * NH, (long)H,
            (long)Nh * H * H, (long)H * H, Nh);

    // 8) qz[(b*N+n)*S + q] = eps + phi_q . z
    compute_qz<<<dim3((Bc * Nh * S) / 8), blk, 0, stream>>>(phi_q, zVec, qzBuf, S, Nh, H);

    // 9) out[b,q,n*H+h] = (phi_q[b,:,n,:] (512x128) @ s[b,n] (128x128)) / qz
    gemm_bf16_wmma<unsigned short, float, float, 2, true, true>
        <<<dim3(H / BN, S / BM, Bc * Nh), blk, 0, stream>>>(
            phi_q, sMat, out, qzBuf, S, H, H,
            (long)NH, 1L, (long)H, 1L, (long)NH, 1L,
            (long)S * NH, (long)H, (long)Nh * H * H, (long)H * H,
            (long)S * NH, (long)H, Nh);
}
